// MoELayer_6605659701906
// MI455X (gfx1250) — matrix operand order repaired and checked by reading, not compiled, not run on hardware
//
#include <hip/hip_runtime.h>
#include <hip/hip_bf16.h>
#include <math.h>

// ---------------- problem constants (match reference setup_inputs) ----------
#define T_TOK 4096          // B*N = 2*2048
#define CDIM  1024          // C
#define FFDIM 4096          // FF
#define NEXP  8             // E
#define TOPK  2
#define TILEM 64            // tokens per block (4 x 16-row WMMA sub-tiles)
#define SUBS  4             // TILEM / 16
#define FFSPL 2             // FF reduction split across blocks (blockIdx.z)
#define SLOTS (T_TOK * TOPK)

typedef __attribute__((ext_vector_type(16))) _Float16 v16h;
typedef __attribute__((ext_vector_type(8)))  _Float16 v8h;
typedef __attribute__((ext_vector_type(8)))  float    v8f;

// ---------------------------------------------------------------------------
// Swizzled LDS tile layout: row-major rows of halves, columns grouped into
// granules of 8 halves; physical granule = (g & ~15) | ((g & 15) ^ (row & 15)).
// Keeps 16B alignment for b128 access and is bank-conflict-free for the
// A-operand fetch pattern (lanes 0-15 = 16 different rows, same logical g).
__device__ __forceinline__ int swz(int row, int g) {
    return (g & ~15) | ((g & 15) ^ (row & 15));
}

// A-operand loader from a swizzled LDS tile (rowStride in halves, multiple of
// 128). Per ISA layout: lanes 0-15 hold M=lane; lane's v16h = two contiguous
// 8-half runs at K = kbase + hi*8 and kbase + 16 + hi*8 (hi = lane>>4).
__device__ __forceinline__ v16h load_a_sw(const _Float16* s, int rowStride,
                                          int kbase, int lane) {
    int m  = lane & 15;
    int hi = lane >> 4;
    int g0 = (kbase >> 3) + hi;       // first 8-half granule
    int g1 = g0 + 2;                  // second run is +16 halves
    const v8h* p0 = (const v8h*)(s + m * rowStride + swz(m, g0) * 8);
    const v8h* p1 = (const v8h*)(s + m * rowStride + swz(m, g1) * 8);
    v8h lo = *p0, up = *p1;
    v16h a;
#pragma unroll
    for (int i = 0; i < 8; ++i) { a[i] = lo[i]; a[8 + i] = up[i]; }
    return a;
}

// B-operand built straight from global f32 weights, no LDS round-trip.
// B layout: lane l holds N=l&15, element i holds K = (l>>4)*8 + (i&7) + 16*(i>>3).
// B[k][n] = W[n][k], so lane reads W[l&15][kbase + K(i)].
__device__ __forceinline__ void ldB(float* tmp, const float* __restrict__ Wrow,
                                    int rowStride, int lane) {
#pragma unroll
    for (int n = 0; n < 16; ++n) tmp[n] = Wrow[(size_t)(lane & 15) * rowStride + (lane >> 4) * 8 + (n & 7) + (n >> 3) * 16];
}
__device__ __forceinline__ v16h cvtB(const float* tmp) {
    v16h b;
#pragma unroll
    for (int n = 0; n < 16; ++n) b[n] = (_Float16)tmp[n];
    return b;
}

__device__ __forceinline__ v8f wmma_f16(v16h a, v16h b, v8f c) {
    return __builtin_amdgcn_wmma_f32_16x16x32_f16(false, a, false, b,
                                                  (short)0, c, false, false);
}

// ---------------------------------------------------------------------------
__global__ void init_kernel(int* counts) {
    if (threadIdx.x < NEXP) counts[threadIdx.x] = 0;
}

// One wave32 per token: router logits, softmax, top-2, histogram.
__global__ __launch_bounds__(256) void router_kernel(
    const float* __restrict__ x, const float* __restrict__ rw,
    int* __restrict__ topk_e, float* __restrict__ topk_w, int* __restrict__ counts) {
    int gid   = blockIdx.x * blockDim.x + threadIdx.x;
    int token = gid >> 5;
    int lane  = gid & 31;
    if (token >= T_TOK) return;
    float acc[NEXP];
#pragma unroll
    for (int e = 0; e < NEXP; ++e) acc[e] = 0.f;
    for (int i = 0; i < CDIM; i += 32) {
        float xv = x[(size_t)token * CDIM + i + lane];
#pragma unroll
        for (int e = 0; e < NEXP; ++e) acc[e] += xv * rw[e * CDIM + i + lane];
    }
#pragma unroll
    for (int e = 0; e < NEXP; ++e)
        for (int off = 16; off > 0; off >>= 1) acc[e] += __shfl_xor(acc[e], off, 32);
    if (lane == 0) {
        float mx = acc[0];
#pragma unroll
        for (int e = 1; e < NEXP; ++e) mx = fmaxf(mx, acc[e]);
        float p[NEXP], s = 0.f;
#pragma unroll
        for (int e = 0; e < NEXP; ++e) { p[e] = __expf(acc[e] - mx); s += p[e]; }
        int e0 = 0;
#pragma unroll
        for (int e = 1; e < NEXP; ++e) if (p[e] > p[e0]) e0 = e;   // ties: lower idx
        int e1 = (e0 == 0) ? 1 : 0;
#pragma unroll
        for (int e = 0; e < NEXP; ++e) if (e != e0 && p[e] > p[e1]) e1 = e;
        float w0 = p[e0] / s, w1 = p[e1] / s;
        float d = w0 + w1 + 1e-9f;
        topk_e[token * 2 + 0] = e0;
        topk_e[token * 2 + 1] = e1;
        topk_w[token * 2 + 0] = w0 / d;
        topk_w[token * 2 + 1] = w1 / d;
        atomicAdd(&counts[e0], 1);
        atomicAdd(&counts[e1], 1);
    }
}

__global__ void scan_kernel(const int* __restrict__ counts,
                            int* __restrict__ offsets, int* __restrict__ cursors) {
    if (threadIdx.x == 0) {
        int run = 0;
        for (int e = 0; e < NEXP; ++e) { offsets[e] = run; cursors[e] = run; run += counts[e]; }
    }
}

__global__ void scatter_kernel(const int* __restrict__ topk_e, int* __restrict__ cursors,
                               int* __restrict__ bucket_tok, int* __restrict__ slot_of) {
    int token = blockIdx.x * blockDim.x + threadIdx.x;
    if (token >= T_TOK) return;
#pragma unroll
    for (int k = 0; k < TOPK; ++k) {
        int e   = topk_e[token * 2 + k];
        int pos = atomicAdd(&cursors[e], 1);
        bucket_tok[pos]          = token;
        slot_of[token * 2 + k]   = pos;
    }
}

// ---------------------------------------------------------------------------
// Fused expert FFN for one (expert, 64-token tile, half of FF):
//   part[z][64,1024] = gelu(X @ W1[e]^T + b1)[:, z*2048:(z+1)*2048] @ W2^T
// 8 waves; wave w owns output cols [w*128, w*128+128).  Four 16-row A
// sub-tiles share every B operand -> 4 WMMAs per 16 weight loads.  B is
// double-buffered (convert-ahead) so post-WMMA VALU never writes B regs.
__global__ __launch_bounds__(256) void ffn_kernel(
    const float* __restrict__ x,
    const float* __restrict__ w1, const float* __restrict__ b1,
    const float* __restrict__ w2, const float* __restrict__ b2,
    const int* __restrict__ offsets, const int* __restrict__ counts,
    const int* __restrict__ bucket_tok, float* __restrict__ out_slots) {
    int tile = blockIdx.x;              // tile varies fastest -> same-expert
    int e    = blockIdx.y;              // blocks adjacent -> weight reuse in L2
    int zh   = blockIdx.z;              // which half of FF this block reduces
    int cnt  = counts[e];
    int off  = offsets[e];
    int rows = cnt - tile * TILEM;
    if (rows <= 0) return;              // block-uniform exit
    if (rows > TILEM) rows = TILEM;

    __shared__ _Float16 sX[TILEM * CDIM];   // 128 KB, swizzled
    __shared__ _Float16 sH[TILEM * 128];    //  16 KB, swizzled

    int tid = threadIdx.x, wv = tid >> 5, lane = tid & 31;

    // ---- stage x tile (f32 -> f16): 4 threads per row, 256 cols each ----
    {
        int r  = tid >> 2;
        int c0 = (tid & 3) * 256;
        int token = (r < rows) ? bucket_tok[off + tile * TILEM + r] : -1;
#pragma unroll
        for (int part = 0; part < 4; ++part) {
            float4 vbuf[16];
            if (token >= 0) {
                const float4* src =
                    (const float4*)(x + (size_t)token * CDIM + c0 + part * 64);
#pragma unroll
                for (int q = 0; q < 16; ++q) vbuf[q] = src[q];
            } else {
#pragma unroll
                for (int q = 0; q < 16; ++q) vbuf[q] = make_float4(0.f, 0.f, 0.f, 0.f);
            }
#pragma unroll
            for (int q = 0; q < 8; ++q) {
                v8h h8;
                h8[0] = (_Float16)vbuf[2 * q].x;     h8[1] = (_Float16)vbuf[2 * q].y;
                h8[2] = (_Float16)vbuf[2 * q].z;     h8[3] = (_Float16)vbuf[2 * q].w;
                h8[4] = (_Float16)vbuf[2 * q + 1].x; h8[5] = (_Float16)vbuf[2 * q + 1].y;
                h8[6] = (_Float16)vbuf[2 * q + 1].z; h8[7] = (_Float16)vbuf[2 * q + 1].w;
                int g = (c0 + part * 64) / 8 + q;    // logical granule
                *(v8h*)(sX + r * CDIM + swz(r, g) * 8) = h8;
            }
        }
    }
    __syncthreads();

    v8f c2[SUBS][8];
#pragma unroll
    for (int s = 0; s < SUBS; ++s)
#pragma unroll
        for (int t = 0; t < 8; ++t) c2[s][t] = (v8f){};

    const float* W1e = w1 + (size_t)e * FFDIM * CDIM;
    const float* W2e = w2 + (size_t)e * CDIM * FFDIM;
    const int ffStart = zh * (FFDIM / FFSPL);
    const int ffEnd   = ffStart + (FFDIM / FFSPL);

    for (int ff0 = ffStart; ff0 < ffEnd; ff0 += 128) {
        // ---- stage 1: wave w computes h[64, ff0+w*16 .. +16) ----
        int ffBase = ff0 + wv * 16;
        const float* base1 = W1e + (size_t)ffBase * CDIM;
        v8f hAcc[SUBS];
#pragma unroll
        for (int s = 0; s < SUBS; ++s) hAcc[s] = (v8f){};
        float tbuf[16];
        ldB(tbuf, base1, CDIM, lane);                 // loads for kc=0
        v16h bcur = cvtB(tbuf);                       // B(kc=0)
        ldB(tbuf, base1 + 32, CDIM, lane);            // loads for kc=1
        for (int kc = 0; kc < CDIM / 32; ++kc) {
            v16h a[SUBS];
#pragma unroll
            for (int s = 0; s < SUBS; ++s)
                a[s] = load_a_sw(sX + s * 16 * CDIM, CDIM, kc * 32, lane);
            v16h bnow = bcur;
            bcur = cvtB(tbuf);                        // B(kc+1), dead at tail
            int kpre = (kc + 2 < CDIM / 32) ? kc + 2 : CDIM / 32 - 1;
            ldB(tbuf, base1 + kpre * 32, CDIM, lane); // loads for kc+2
#pragma unroll
            for (int s = 0; s < SUBS; ++s) hAcc[s] = wmma_f16(a[s], bnow, hAcc[s]);
        }
        // bias + exact GELU, write h tile to swizzled LDS (f16)
        {
            float bb = b1[e * FFDIM + ffBase + (lane & 15)];
            int n   = lane & 15;
            int mhi = (lane >> 4) * 8;
            int cc  = wv * 16 + n;            // column within 128-wide chunk
            int g   = cc >> 3;                // logical granule of that column
#pragma unroll
            for (int s = 0; s < SUBS; ++s) {
#pragma unroll
                for (int r = 0; r < 8; ++r) {
                    int row = s * 16 + mhi + r;
                    float v = hAcc[s][r] + bb;
                    v = 0.5f * v * (1.f + erff(v * 0.70710678118f));
                    sH[row * 128 + swz(row, g) * 8 + (cc & 7)] = (_Float16)v;
                }
            }
        }
        __syncthreads();

        // ---- stage 2: accumulate out[64, wv*128 .. +128) over this FF chunk
        // K-outer / N-inner: A operands loaded once per K-chunk (8x fewer DS).
        const float* base2 = W2e + (size_t)(wv * 128) * FFDIM + ff0;
        float ubuf[16];
        v16h aS[SUBS];
        // it = kc*8 + t;  addr(it) = base2 + t*16*FFDIM + kc*32
        ldB(ubuf, base2, FFDIM, lane);                // it=0
        v16h b2cur = cvtB(ubuf);
        ldB(ubuf, base2 + (size_t)16 * FFDIM, FFDIM, lane);   // it=1
#pragma unroll
        for (int it = 0; it < 32; ++it) {
            int kc = it >> 3, t = it & 7;
            if (t == 0) {
#pragma unroll
                for (int s = 0; s < SUBS; ++s)
                    aS[s] = load_a_sw(sH + s * 16 * 128, 128, kc * 32, lane);
            }
            v16h bnow = b2cur;
            if (it + 1 < 32) {
                b2cur = cvtB(ubuf);
                int in2 = (it + 2 < 32) ? it + 2 : 31;
                ldB(ubuf, base2 + (size_t)((in2 & 7) * 16) * FFDIM + (in2 >> 3) * 32,
                    FFDIM, lane);
            }
#pragma unroll
            for (int s = 0; s < SUBS; ++s) c2[s][t] = wmma_f16(aS[s], bnow, c2[s][t]);
        }
        __syncthreads();
    }

    // ---- writeout partials: +b2 only from z-half 0 (combine sums halves) ----
    {
        float* part = out_slots + (size_t)zh * SLOTS * CDIM;
        int n  = lane & 15;
        int hi = lane >> 4;
#pragma unroll
        for (int t = 0; t < 8; ++t) {
            int col   = wv * 128 + t * 16 + n;
            float bb2 = (zh == 0) ? b2[e * CDIM + col] : 0.f;
#pragma unroll
            for (int s = 0; s < SUBS; ++s) {
#pragma unroll
                for (int r = 0; r < 8; ++r) {
                    int m = s * 16 + r + 8 * hi;
                    if (m < rows) {
                        int slot = off + tile * TILEM + m;
                        part[(size_t)slot * CDIM + col] = c2[s][t][r] + bb2;
                    }
                }
            }
        }
    }
}

// out[token, col] = w0*(p0[s0]+p1[s0]) + w1*(p0[s1]+p1[s1])   (deterministic)
__global__ __launch_bounds__(256) void combine_kernel(
    const float* __restrict__ out_slots, const int* __restrict__ slot_of,
    const float* __restrict__ topk_w, float* __restrict__ out) {
    int i = blockIdx.x * blockDim.x + threadIdx.x;
    if (i >= T_TOK * CDIM) return;
    int token = i >> 10;            // / CDIM
    int col   = i & (CDIM - 1);
    const float* p0 = out_slots;
    const float* p1 = out_slots + (size_t)SLOTS * CDIM;
    int s0 = slot_of[token * 2 + 0];
    int s1 = slot_of[token * 2 + 1];
    float w0 = topk_w[token * 2 + 0];
    float w1 = topk_w[token * 2 + 1];
    float o0 = p0[(size_t)s0 * CDIM + col] + p1[(size_t)s0 * CDIM + col];
    float o1 = p0[(size_t)s1 * CDIM + col] + p1[(size_t)s1 * CDIM + col];
    out[i] = w0 * o0 + w1 * o1;
}

// ---------------------------------------------------------------------------
extern "C" void kernel_launch(void* const* d_in, const int* in_sizes, int n_in,
                              void* d_out, int out_size, void* d_ws, size_t ws_size,
                              hipStream_t stream) {
    const float* x        = (const float*)d_in[0];
    const float* router_w = (const float*)d_in[1];
    const float* w1       = (const float*)d_in[2];
    const float* b1       = (const float*)d_in[3];
    const float* w2       = (const float*)d_in[4];
    const float* b2       = (const float*)d_in[5];
    float* out            = (float*)d_out;

    char* ws = (char*)d_ws;
    size_t p = 0;
    auto take = [&](size_t bytes) -> char* {
        char* q = ws + p;
        p = (p + bytes + 255) & ~(size_t)255;
        return q;
    };
    int*   counts     = (int*)take(NEXP * sizeof(int));
    int*   offsets    = (int*)take(NEXP * sizeof(int));
    int*   cursors    = (int*)take(NEXP * sizeof(int));
    int*   topk_e     = (int*)take((size_t)SLOTS * sizeof(int));
    float* topk_w     = (float*)take((size_t)SLOTS * sizeof(float));
    int*   slot_of    = (int*)take((size_t)SLOTS * sizeof(int));
    int*   bucket_tok = (int*)take((size_t)SLOTS * sizeof(int));
    float* out_slots  = (float*)take((size_t)FFSPL * SLOTS * CDIM * sizeof(float)); // 64 MB

    init_kernel<<<1, 32, 0, stream>>>(counts);
    router_kernel<<<(T_TOK * 32) / 256, 256, 0, stream>>>(x, router_w, topk_e, topk_w, counts);
    scan_kernel<<<1, 32, 0, stream>>>(counts, offsets, cursors);
    scatter_kernel<<<(T_TOK + 255) / 256, 256, 0, stream>>>(topk_e, cursors, bucket_tok, slot_of);
    ffn_kernel<<<dim3((SLOTS + TILEM - 1) / TILEM, NEXP, FFSPL), 256, 0, stream>>>(
        x, w1, b1, w2, b2, offsets, counts, bucket_tok, out_slots);
    combine_kernel<<<(T_TOK * CDIM + 255) / 256, 256, 0, stream>>>(
        out_slots, slot_of, topk_w, out);
}
